// CoAttention_33861522162608
// MI455X (gfx1250) — matrix operand order repaired and checked by reading, not compiled, not run on hardware
//
#include <hip/hip_runtime.h>
#include <hip/hip_bf16.h>

#define B_  64
#define CL_ 2048
#define QL_ 256
#define H_  512

typedef __attribute__((ext_vector_type(16))) __bf16          v16bf;
typedef __attribute__((ext_vector_type(16))) unsigned short  v16u;
typedef __attribute__((ext_vector_type(8)))  unsigned short  v8u;
typedef __attribute__((ext_vector_type(8)))  float           v8f;

#define KC   64  // K chunk for GEMM2 (vmem path)
#define LDT  72  // padded LDS row stride for KC=64 (144 B, multiple of 16)
#define KC2  32  // K chunk for TDM double-buffered GEMMs
#define LDT2 40  // padded LDS row stride for KC2=32 (80 B = 64 B data + 16 B pad)

__device__ __forceinline__ unsigned short f2bf(float f) {
  unsigned u = __float_as_uint(f);
  u += 0x7FFFu + ((u >> 16) & 1u);          // round-to-nearest-even
  return (unsigned short)(u >> 16);
}

// A fragment (16x32 bf16, MxK): lane half h -> K = h*8 + [0..7] and 16 + h*8 + [0..7]
__device__ __forceinline__ v16bf frag_a(const unsigned short* ldsRow, int half, int kk) {
  v8u lo = *(const v8u*)(ldsRow + kk + half * 8);
  v8u hi = *(const v8u*)(ldsRow + kk + 16 + half * 8);
  v16u c = __builtin_shufflevector(lo, hi, 0,1,2,3,4,5,6,7,8,9,10,11,12,13,14,15);
  return __builtin_bit_cast(v16bf, c);
}

// B fragment (32x16 bf16, KxN): lane half h -> K = h*8 + [0..7] and 16 + h*8 + [0..7]
__device__ __forceinline__ v16bf frag_b(const unsigned short* ldsRow, int half, int kk) {
  v8u lo = *(const v8u*)(ldsRow + kk + half * 8);
  v8u hi = *(const v8u*)(ldsRow + kk + 16 + half * 8);
  v16u c = __builtin_shufflevector(lo, hi, 0,1,2,3,4,5,6,7,8,9,10,11,12,13,14,15);
  return __builtin_bit_cast(v16bf, c);
}

// 8 N-subtiles of WMMA with B-fragment load pipelined one subtile ahead.
template<int KCOUNT, int STRIDE>
__device__ __forceinline__ void mma_chunk(const unsigned short* ldsA, const unsigned short* ldsB,
                                          int w, int l16, int half, v8f acc[8]) {
#pragma unroll
  for (int kk = 0; kk < KCOUNT; kk += 32) {
    v16bf a  = frag_a(ldsA + (w * 16 + l16) * STRIDE, half, kk);
    v16bf bc = frag_b(ldsB + l16 * STRIDE, half, kk);
#pragma unroll
    for (int s = 0; s < 8; ++s) {
      v16bf bn = bc;
      if (s < 7) bn = frag_b(ldsB + ((s + 1) * 16 + l16) * STRIDE, half, kk);
      acc[s] = __builtin_amdgcn_wmma_f32_16x16x32_bf16(false, a, false, bc,
                                                       (short)0, acc[s], false, false);
      bc = bn;
    }
  }
}

// ---------------- TDM tile load (128 rows x 32 bf16, padded to 80 B rows) ----------------

#if __has_builtin(__builtin_amdgcn_tensor_load_to_lds)
#define USE_TDM 1
typedef __attribute__((ext_vector_type(4))) unsigned int v4u32;
typedef __attribute__((ext_vector_type(8))) int          v8i32;
typedef __attribute__((ext_vector_type(4))) int          v4i32;

__device__ __forceinline__ void tdm_load_tile32(const unsigned short* g, unsigned ldsByteOff,
                                                int gstrideElems) {
  unsigned long long ga = (unsigned long long)(size_t)g;
  v4u32 g0;
  g0[0] = 1u;                                              // count=1 (valid user descriptor)
  g0[1] = ldsByteOff;                                      // lds_addr (bytes)
  g0[2] = (unsigned)(ga & 0xFFFFFFFFull);                  // global_addr[31:0]
  g0[3] = (unsigned)((ga >> 32) & 0x01FFFFFFull) | (2u << 30); // global_addr[56:32], type=2
  v8i32 g1;
  // data_size=1 (2B), pad_enable, pad_interval=3 (16 DWORDs = 64 B row), pad_amount=3 (4 DWORDs = 16 B)
  g1[0] = (int)((1u << 16) | (1u << 20) | (3u << 22) | (3u << 25));
  unsigned td0 = (unsigned)gstrideElems;                   // tensor_dim0 = full row length
  unsigned td1 = 128u;                                     // tensor_dim1 = tile rows
  g1[1] = (int)((td0 & 0xFFFFu) << 16);                    // [63:48] = tensor_dim0 lo16
  g1[2] = (int)(((td0 >> 16) & 0xFFFFu) | ((td1 & 0xFFFFu) << 16)); // dim0 hi16 | dim1 lo16
  g1[3] = (int)(((td1 >> 16) & 0xFFFFu) | (32u << 16));    // dim1 hi16 | tile_dim0=32
  g1[4] = (int)(128u);                                     // tile_dim1=128, tile_dim2=0
  g1[5] = (int)gstrideElems;                               // tensor_dim0_stride lo32 (elems)
  g1[6] = 0;                                               // stride hi16 | dim1_stride lo16
  g1[7] = 0;
  v4i32 z4 = {0, 0, 0, 0};
  v8i32 z8 = {0, 0, 0, 0, 0, 0, 0, 0};
  __builtin_amdgcn_tensor_load_to_lds(g0, g1, z4, z4, z8, 0);
}
#else
#define USE_TDM 0
#endif

// Fallback / GEMM2 fills ------------------------------------------------------

// Copy a 128 x 32 bf16 tile (row-major) into padded LDS (stride LDT2). 256 threads.
__device__ __forceinline__ void fill_direct2(unsigned short* lds, const unsigned short* g, int gstride) {
  int t = threadIdx.x;
#pragma unroll
  for (int i = 0; i < 2; ++i) {
    int idx = t + i * 256;          // 512 uint4's
    int r   = idx >> 2;
    int cc  = (idx & 3) << 3;
    *(uint4*)(lds + r * LDT2 + cc) = *(const uint4*)(g + (size_t)r * gstride + cc);
  }
}

// LDS[mm][kk] = g[kk*gstride + mm] for a 128(mm) x 64(kk) tile (transposed load, stride LDT).
__device__ __forceinline__ void fill_transpose(unsigned short* lds, const unsigned short* g, int gstride) {
  int t = threadIdx.x;
#pragma unroll
  for (int i = 0; i < 4; ++i) {
    int idx = t + i * 256;
    int kk  = idx >> 4;             // 0..63
    int mm0 = (idx & 15) << 3;      // 0..120 step 8
    uint4 v = *(const uint4*)(g + (size_t)kk * gstride + mm0);
    const unsigned short* p = (const unsigned short*)&v;
#pragma unroll
    for (int j = 0; j < 8; ++j) lds[(mm0 + j) * LDT + kk] = p[j];
  }
}

// ---------------- conversion kernels ----------------

__global__ void cvt_bf16_kernel(const float* __restrict__ in, unsigned short* __restrict__ out, int n4) {
  int i = blockIdx.x * 256 + threadIdx.x;
  if (i >= n4) return;
  float4 v = ((const float4*)in)[i];
  uint2 u;
  u.x = (unsigned)f2bf(v.x) | ((unsigned)f2bf(v.y) << 16);
  u.y = (unsigned)f2bf(v.z) | ((unsigned)f2bf(v.w) << 16);
  *(uint2*)(out + (size_t)i * 4) = u;
}

// concat[b, h, ql] = bf16(q[b, ql, h]) for h < H (rows 0..H-1 of concatenated)
__global__ void qT_kernel(const float* __restrict__ q, unsigned short* __restrict__ concatbf) {
  int idx = blockIdx.x * 256 + threadIdx.x;   // B*H*QL
  if (idx >= B_ * H_ * QL_) return;
  int ql = idx & (QL_ - 1);
  int t  = idx >> 8;
  int h  = t & (H_ - 1);
  int b  = t >> 9;
  concatbf[((size_t)(b * 2 * H_ + h)) * QL_ + ql] = f2bf(q[((size_t)b * QL_ + ql) * H_ + h]);
}

// ---------------- GEMM1: L = c @ q^T  [CL x QL], TDM double-buffered ----------------

__global__ void gemm_L_kernel(const unsigned short* __restrict__ cbf,
                              const unsigned short* __restrict__ qbf,
                              float* __restrict__ Lout) {
  const int nt = QL_ / 128, mt = CL_ / 128;
  int bid = blockIdx.x;
  int b  = bid / (mt * nt);
  int r  = bid % (mt * nt);
  int m0 = (r / nt) * 128;
  int n0 = (r % nt) * 128;
  __shared__ alignas(16) unsigned short ldsA[2][128 * LDT2];
  __shared__ alignas(16) unsigned short ldsB[2][128 * LDT2];
  const unsigned short* Ag = cbf + ((size_t)b * CL_ + m0) * H_;
  const unsigned short* Bg = qbf + ((size_t)b * QL_ + n0) * H_;
  int w = threadIdx.x >> 5, lane = threadIdx.x & 31;
  int l16 = lane & 15, half = lane >> 4;
  v8f acc[8] = {};
  const int NCH = H_ / KC2;
#if USE_TDM
  bool issuer = (threadIdx.x < 32);
  unsigned offA[2] = { (unsigned)(size_t)(void*)&ldsA[0][0], (unsigned)(size_t)(void*)&ldsA[1][0] };
  unsigned offB[2] = { (unsigned)(size_t)(void*)&ldsB[0][0], (unsigned)(size_t)(void*)&ldsB[1][0] };
  if (issuer) {
    tdm_load_tile32(Ag, offA[0], H_);
    tdm_load_tile32(Bg, offB[0], H_);
  }
  for (int ch = 0; ch < NCH; ++ch) {
    if (issuer) __builtin_amdgcn_s_wait_tensorcnt(0);
    __syncthreads();                    // data for buf ch&1 visible; prior reads of other buf done
    int nx = ch + 1;
    if (issuer && nx < NCH) {
      tdm_load_tile32(Ag + nx * KC2, offA[nx & 1], H_);
      tdm_load_tile32(Bg + nx * KC2, offB[nx & 1], H_);
    }
    mma_chunk<KC2, LDT2>(ldsA[ch & 1], ldsB[ch & 1], w, l16, half, acc);
  }
#else
  for (int ch = 0; ch < NCH; ++ch) {
    __syncthreads();
    fill_direct2(ldsA[0], Ag + ch * KC2, H_);
    fill_direct2(ldsB[0], Bg + ch * KC2, H_);
    __syncthreads();
    mma_chunk<KC2, LDT2>(ldsA[0], ldsB[0], w, l16, half, acc);
  }
#endif
  float* Lb = Lout + (size_t)b * CL_ * QL_;
#pragma unroll
  for (int s = 0; s < 8; ++s) {
    int n = n0 + s * 16 + l16;
#pragma unroll
    for (int rr = 0; rr < 8; ++rr) {
      int m = m0 + w * 16 + half * 8 + rr;
      Lb[(size_t)m * QL_ + n] = acc[s][rr];
    }
  }
}

// ---------------- softmax over q axis (per c-row), q_mask ----------------

__global__ void softmax_ac_kernel(const float* __restrict__ L, const int* __restrict__ qmask,
                                  unsigned short* __restrict__ acbf) {
  int row = blockIdx.x;             // b*CL + c
  int b   = row >> 11;              // / CL_
  int ql  = threadIdx.x;
  float x = (qmask[b * QL_ + ql] != 0) ? L[(size_t)row * QL_ + ql] : -1e30f;
  __shared__ float red[256];
  red[ql] = x; __syncthreads();
  for (int s = 128; s > 0; s >>= 1) { if (ql < s) red[ql] = fmaxf(red[ql], red[ql + s]); __syncthreads(); }
  float mx = red[0]; __syncthreads();
  float e = __expf(x - mx);
  red[ql] = e; __syncthreads();
  for (int s = 128; s > 0; s >>= 1) { if (ql < s) red[ql] += red[ql + s]; __syncthreads(); }
  float sum = red[0];
  acbf[(size_t)row * QL_ + ql] = f2bf(e / sum);
}

// ---------------- softmax over c axis (per q-column), c_mask, online ----------------

__global__ void softmax_aq_kernel(const float* __restrict__ L, const int* __restrict__ cmask,
                                  unsigned short* __restrict__ aqbf) {
  int b  = blockIdx.x >> 2;
  int cg = blockIdx.x & 3;
  int tx = threadIdx.x;
  int col = (cg << 6) + (tx & 63);  // ql
  int seg = tx >> 6;                // 0..3
  const int rps = CL_ / 4;          // 512 rows per segment
  const float* Lb = L + (size_t)b * CL_ * QL_;
  const int* cm = cmask + b * CL_;
  float m = -1e30f, ssum = 0.f;
  int c0 = seg * rps;
  for (int c = c0; c < c0 + rps; ++c) {
    float x = (cm[c] != 0) ? Lb[(size_t)c * QL_ + col] : -1e30f;
    if (x > m) { ssum = ssum * __expf(m - x) + 1.f; m = x; }
    else        ssum += __expf(x - m);
  }
  __shared__ float sm[256], ss[256];
  sm[tx] = m; ss[tx] = ssum;
  __syncthreads();
  int base = tx & 63;
  float M = fmaxf(fmaxf(sm[base], sm[base + 64]), fmaxf(sm[base + 128], sm[base + 192]));
  float S = ss[base]       * __expf(sm[base]       - M) + ss[base + 64]  * __expf(sm[base + 64]  - M)
          + ss[base + 128] * __expf(sm[base + 128] - M) + ss[base + 192] * __expf(sm[base + 192] - M);
  float invS = 1.f / S;
  unsigned short* aqb = aqbf + (size_t)b * CL_ * QL_;
  for (int c = c0; c < c0 + rps; ++c) {
    float x = (cm[c] != 0) ? Lb[(size_t)c * QL_ + col] : -1e30f;
    aqb[(size_t)c * QL_ + col] = f2bf(__expf(x - M) * invS);
  }
}

// ---------------- GEMM2: Cq = c^T @ aq  [H x QL] -> concat rows H..2H-1 (bf16) ----------------

__global__ void gemm_Cq_kernel(const unsigned short* __restrict__ cbf,
                               const unsigned short* __restrict__ aqbf,
                               unsigned short* __restrict__ concatbf) {
  const int nt = QL_ / 128, mt = H_ / 128;
  int bid = blockIdx.x;
  int b  = bid / (mt * nt);
  int r  = bid % (mt * nt);
  int m0 = (r / nt) * 128;    // over H
  int n0 = (r % nt) * 128;    // over QL
  __shared__ alignas(16) unsigned short ldsA[128 * LDT];
  __shared__ alignas(16) unsigned short ldsB[128 * LDT];
  int w = threadIdx.x >> 5, lane = threadIdx.x & 31;
  int l16 = lane & 15, half = lane >> 4;
  v8f acc[8] = {};
  for (int k0 = 0; k0 < CL_; k0 += KC) {
    __syncthreads();
    fill_transpose(ldsA, cbf  + ((size_t)b * CL_ + k0) * H_  + m0, H_);
    fill_transpose(ldsB, aqbf + ((size_t)b * CL_ + k0) * QL_ + n0, QL_);
    __syncthreads();
    mma_chunk<KC, LDT>(ldsA, ldsB, w, l16, half, acc);
  }
  unsigned short* Cb = concatbf + ((size_t)b * 2 * H_ + H_) * QL_;
#pragma unroll
  for (int s = 0; s < 8; ++s) {
    int n = n0 + s * 16 + l16;
#pragma unroll
    for (int rr = 0; rr < 8; ++rr) {
      int m = m0 + w * 16 + half * 8 + rr;
      Cb[(size_t)m * QL_ + n] = f2bf(acc[s][rr]);
    }
  }
}

// ---------------- GEMM3: Cc = concat @ ac^T -> out[:, :, H:3H], TDM double-buffered ----------------

__global__ void gemm_out_kernel(const unsigned short* __restrict__ concatbf,
                                const unsigned short* __restrict__ acbf,
                                float* __restrict__ out) {
  const int nt = CL_ / 128, mt = (2 * H_) / 128;
  int bid = blockIdx.x;
  int b  = bid / (mt * nt);
  int r  = bid % (mt * nt);
  int m0 = (r / nt) * 128;    // over 2H
  int n0 = (r % nt) * 128;    // over CL
  __shared__ alignas(16) unsigned short ldsA[2][128 * LDT2];
  __shared__ alignas(16) unsigned short ldsB[2][128 * LDT2];
  const unsigned short* Ag = concatbf + ((size_t)b * 2 * H_ + m0) * QL_;
  const unsigned short* Bg = acbf     + ((size_t)b * CL_    + n0) * QL_;
  int w = threadIdx.x >> 5, lane = threadIdx.x & 31;
  int l16 = lane & 15, half = lane >> 4;
  v8f acc[8] = {};
  const int NCH = QL_ / KC2;
#if USE_TDM
  bool issuer = (threadIdx.x < 32);
  unsigned offA[2] = { (unsigned)(size_t)(void*)&ldsA[0][0], (unsigned)(size_t)(void*)&ldsA[1][0] };
  unsigned offB[2] = { (unsigned)(size_t)(void*)&ldsB[0][0], (unsigned)(size_t)(void*)&ldsB[1][0] };
  if (issuer) {
    tdm_load_tile32(Ag, offA[0], QL_);
    tdm_load_tile32(Bg, offB[0], QL_);
  }
  for (int ch = 0; ch < NCH; ++ch) {
    if (issuer) __builtin_amdgcn_s_wait_tensorcnt(0);
    __syncthreads();
    int nx = ch + 1;
    if (issuer && nx < NCH) {
      tdm_load_tile32(Ag + nx * KC2, offA[nx & 1], QL_);
      tdm_load_tile32(Bg + nx * KC2, offB[nx & 1], QL_);
    }
    mma_chunk<KC2, LDT2>(ldsA[ch & 1], ldsB[ch & 1], w, l16, half, acc);
  }
#else
  for (int ch = 0; ch < NCH; ++ch) {
    __syncthreads();
    fill_direct2(ldsA[0], Ag + ch * KC2, QL_);
    fill_direct2(ldsB[0], Bg + ch * KC2, QL_);
    __syncthreads();
    mma_chunk<KC2, LDT2>(ldsA[0], ldsB[0], w, l16, half, acc);
  }
#endif
#pragma unroll
  for (int s = 0; s < 8; ++s) {
    int n = n0 + s * 16 + l16;              // c index
    float* row = out + ((size_t)b * CL_ + n) * (3 * H_) + H_ + m0 + w * 16 + half * 8;
    float4 v0 = make_float4(acc[s][0], acc[s][1], acc[s][2], acc[s][3]);
    float4 v1 = make_float4(acc[s][4], acc[s][5], acc[s][6], acc[s][7]);
    *(float4*)row       = v0;
    *(float4*)(row + 4) = v1;
  }
}

// ---------------- copy c into out[:, :, 0:H] ----------------

__global__ void copy_c_kernel(const float* __restrict__ c, float* __restrict__ out) {
  int i = blockIdx.x * 256 + threadIdx.x;   // over B*CL*H/4
  int row = i >> 7;                          // / (H_/4)
  int hc  = i & 127;
  ((float4*)out)[(size_t)row * (3 * H_ / 4) + hc] = ((const float4*)c)[(size_t)row * 128 + hc];
}

extern "C" void kernel_launch(void* const* d_in, const int* in_sizes, int n_in,
                              void* d_out, int out_size, void* d_ws, size_t ws_size,
                              hipStream_t stream) {
  const float* c     = (const float*)d_in[0];
  const float* q     = (const float*)d_in[1];
  const int*   cmask = (const int*)d_in[2];
  const int*   qmask = (const int*)d_in[3];
  float* out = (float*)d_out;

  char* w = (char*)d_ws;
  unsigned short* cbf      = (unsigned short*)w;  w += (size_t)B_ * CL_ * H_ * 2;
  unsigned short* qbf      = (unsigned short*)w;  w += (size_t)B_ * QL_ * H_ * 2;
  float*          Lbuf     = (float*)w;           w += (size_t)B_ * CL_ * QL_ * 4;
  unsigned short* acbf     = (unsigned short*)w;  w += (size_t)B_ * CL_ * QL_ * 2;
  unsigned short* aqbf     = (unsigned short*)w;  w += (size_t)B_ * CL_ * QL_ * 2;
  unsigned short* concatbf = (unsigned short*)w;  w += (size_t)B_ * 2 * H_ * QL_ * 2;

  cvt_bf16_kernel<<<(B_ * CL_ * H_ / 4) / 256, 256, 0, stream>>>(c, cbf, B_ * CL_ * H_ / 4);
  cvt_bf16_kernel<<<(B_ * QL_ * H_ / 4) / 256, 256, 0, stream>>>(q, qbf, B_ * QL_ * H_ / 4);
  qT_kernel<<<(B_ * H_ * QL_) / 256, 256, 0, stream>>>(q, concatbf);

  gemm_L_kernel<<<B_ * (CL_ / 128) * (QL_ / 128), 256, 0, stream>>>(cbf, qbf, Lbuf);

  softmax_ac_kernel<<<B_ * CL_, 256, 0, stream>>>(Lbuf, qmask, acbf);
  softmax_aq_kernel<<<B_ * (QL_ / 64), 256, 0, stream>>>(Lbuf, cmask, aqbf);

  gemm_Cq_kernel<<<B_ * (H_ / 128) * (QL_ / 128), 256, 0, stream>>>(cbf, aqbf, concatbf);
  gemm_out_kernel<<<B_ * (2 * H_ / 128) * (CL_ / 128), 256, 0, stream>>>(concatbf, acbf, out);

  copy_c_kernel<<<(B_ * CL_ * H_ / 4) / 256, 256, 0, stream>>>(c, out);
}
